// PointNet2Discriminator_14078902796618
// MI455X (gfx1250) — matrix operand order repaired and checked by reading, not compiled, not run on hardware
//
#include <hip/hip_runtime.h>
#include <hip/hip_bf16.h>

// ---------------------------------------------------------------------------
// CDNA5 WMMA types (wave32): v_wmma_f32_16x16x32_f16
// ---------------------------------------------------------------------------
typedef _Float16 half16 __attribute__((ext_vector_type(16)));
typedef _Float16 half8v __attribute__((ext_vector_type(8)));
typedef float    float8 __attribute__((ext_vector_type(8)));

// A fragment (16x32 MxK f16): lane {half=lane>>4, m=lane&15} holds (this code's order, same as B)
// K in [16h, 16h+15] contiguous -> a[0..15].  Two 16-byte loads.
__device__ __forceinline__ half16 load_a_frag(const _Float16* __restrict__ p) {
  const half8v lo = *(const half8v*)(p);
  const half8v hi = *(const half8v*)(p + 8);
  half16 a;
#pragma unroll
  for (int i = 0; i < 8; ++i) { a[i] = lo[i]; a[8 + i] = hi[i]; }
  return a;
}

// B fragment (32x16 KxN f16) from transposed weights Wt[N][Kp]:
// lane {half, n=lane&15} holds K in [16h, 16h+15] contiguous (same order as A) -> b[0..15].
__device__ __forceinline__ half16 load_b_frag(const _Float16* __restrict__ p) {
  const half8v lo = *(const half8v*)(p);
  const half8v hi = *(const half8v*)(p + 8);
  half16 b;
#pragma unroll
  for (int i = 0; i < 8; ++i) { b[i] = lo[i]; b[8 + i] = hi[i]; }
  return b;
}

// ---------------------------------------------------------------------------
// GEMM: C[M,N] = relu(A[M,K] @ W[K,N] + bias).
// A f16 row-major with padded lda (multiple of 32, zero-padded K tail).
// Wt f16 transposed [N, Kp], Kp = ceil32(K), zero-padded.
// Preconditions (guaranteed by host): M % 16 == 0, N % 64 == 0.
// One 16x64 strip per wave (A fragment reused across 4 WMMAs); 8 waves/block.
// ---------------------------------------------------------------------------
__global__ void gemm_wmma_f16(const _Float16* __restrict__ A, int lda,
                              const _Float16* __restrict__ Wt,
                              const float* __restrict__ bias,
                              _Float16* __restrict__ C, int ldc,
                              int M, int N, int Kp) {
  const int lane = threadIdx.x & 31;
  const int wave = threadIdx.x >> 5;
  const int nstrips = N >> 6;
  const int strip = blockIdx.x * (blockDim.x >> 5) + wave;
  if (strip >= (M >> 4) * nstrips) return;
  const int mt = strip / nstrips;
  const int nc0 = (strip - mt * nstrips) << 6;
  const int hf = lane >> 4;
  const int ln = lane & 15;

  const _Float16* Ap  = A + (size_t)(mt * 16 + ln) * lda + 16 * hf;
  const _Float16* Wp0 = Wt + (size_t)(nc0 + 0 * 16 + ln) * Kp + 16 * hf;
  const _Float16* Wp1 = Wt + (size_t)(nc0 + 1 * 16 + ln) * Kp + 16 * hf;
  const _Float16* Wp2 = Wt + (size_t)(nc0 + 2 * 16 + ln) * Kp + 16 * hf;
  const _Float16* Wp3 = Wt + (size_t)(nc0 + 3 * 16 + ln) * Kp + 16 * hf;

  float8 c0 = {}, c1 = {}, c2 = {}, c3 = {};
  for (int kb = 0; kb < Kp; kb += 32) {
    const half16 a  = load_a_frag(Ap + kb);
    const half16 b0 = load_b_frag(Wp0 + kb);
    const half16 b1 = load_b_frag(Wp1 + kb);
    const half16 b2 = load_b_frag(Wp2 + kb);
    const half16 b3 = load_b_frag(Wp3 + kb);
    c0 = __builtin_amdgcn_wmma_f32_16x16x32_f16(false, a, false, b0, (short)0, c0, false, false);
    c1 = __builtin_amdgcn_wmma_f32_16x16x32_f16(false, a, false, b1, (short)0, c1, false, false);
    c2 = __builtin_amdgcn_wmma_f32_16x16x32_f16(false, a, false, b2, (short)0, c2, false, false);
    c3 = __builtin_amdgcn_wmma_f32_16x16x32_f16(false, a, false, b3, (short)0, c3, false, false);
  }
  float8 acc[4] = {c0, c1, c2, c3};
#pragma unroll
  for (int j = 0; j < 4; ++j) {
    const int col = nc0 + j * 16 + ln;
    const float bc = bias[col];
#pragma unroll
    for (int v = 0; v < 8; ++v) {
      const int row = mt * 16 + v + 8 * hf;
      C[(size_t)row * ldc + col] = (_Float16)fmaxf(acc[j][v] + bc, 0.f);
    }
  }
}

// ---------------------------------------------------------------------------
// GEMM + fused segmented max-pool.  Post-ReLU values >= 0 -> int atomicMax on
// raw float bits is a correct float max (pool pre-zeroed).
// ---------------------------------------------------------------------------
__global__ void gemm_wmma_maxpool(const _Float16* __restrict__ A, int lda,
                                  const _Float16* __restrict__ Wt,
                                  const float* __restrict__ bias,
                                  float* __restrict__ pool,
                                  int M, int N, int Kp, int rowsPerSeg) {
  const int lane = threadIdx.x & 31;
  const int wave = threadIdx.x >> 5;
  const int nstrips = N >> 6;
  const int strip = blockIdx.x * (blockDim.x >> 5) + wave;
  if (strip >= (M >> 4) * nstrips) return;
  const int mt = strip / nstrips;
  const int nc0 = (strip - mt * nstrips) << 6;
  const int hf = lane >> 4;
  const int ln = lane & 15;

  const _Float16* Ap  = A + (size_t)(mt * 16 + ln) * lda + 16 * hf;
  const _Float16* Wp0 = Wt + (size_t)(nc0 + 0 * 16 + ln) * Kp + 16 * hf;
  const _Float16* Wp1 = Wt + (size_t)(nc0 + 1 * 16 + ln) * Kp + 16 * hf;
  const _Float16* Wp2 = Wt + (size_t)(nc0 + 2 * 16 + ln) * Kp + 16 * hf;
  const _Float16* Wp3 = Wt + (size_t)(nc0 + 3 * 16 + ln) * Kp + 16 * hf;

  float8 c0 = {}, c1 = {}, c2 = {}, c3 = {};
  for (int kb = 0; kb < Kp; kb += 32) {
    const half16 a  = load_a_frag(Ap + kb);
    const half16 b0 = load_b_frag(Wp0 + kb);
    const half16 b1 = load_b_frag(Wp1 + kb);
    const half16 b2 = load_b_frag(Wp2 + kb);
    const half16 b3 = load_b_frag(Wp3 + kb);
    c0 = __builtin_amdgcn_wmma_f32_16x16x32_f16(false, a, false, b0, (short)0, c0, false, false);
    c1 = __builtin_amdgcn_wmma_f32_16x16x32_f16(false, a, false, b1, (short)0, c1, false, false);
    c2 = __builtin_amdgcn_wmma_f32_16x16x32_f16(false, a, false, b2, (short)0, c2, false, false);
    c3 = __builtin_amdgcn_wmma_f32_16x16x32_f16(false, a, false, b3, (short)0, c3, false, false);
  }
  float8 acc[4] = {c0, c1, c2, c3};
#pragma unroll
  for (int j = 0; j < 4; ++j) {
    const int col = nc0 + j * 16 + ln;
    const float bc = bias[col];
#pragma unroll
    for (int v = 0; v < 8; ++v) {
      const int row = mt * 16 + v + 8 * hf;
      const float val = fmaxf(acc[j][v] + bc, 0.f);
      atomicMax((int*)&pool[(size_t)(row / rowsPerSeg) * N + col],
                __float_as_int(val));
    }
  }
}

// ---------------------------------------------------------------------------
// Weight convert+transpose: W f32 [K,N] -> Wt f16 [N,Kp], zero-padded K tail.
// ---------------------------------------------------------------------------
__global__ void weight_to_f16t(const float* __restrict__ W, _Float16* __restrict__ Wt,
                               int K, int N, int Kp) {
  const int total = N * Kp;
  for (int t = blockIdx.x * blockDim.x + threadIdx.x; t < total;
       t += gridDim.x * blockDim.x) {
    const int n = t / Kp;
    const int k = t - n * Kp;
    Wt[t] = (k < K) ? (_Float16)W[(size_t)k * N + n] : (_Float16)0.f;
  }
}

// ---------------------------------------------------------------------------
// Small utility kernels
// ---------------------------------------------------------------------------
__global__ void zero_f32(float* p, int n) {
  for (int t = blockIdx.x * blockDim.x + threadIdx.x; t < n;
       t += gridDim.x * blockDim.x)
    p[t] = 0.f;
}

__global__ void f32_to_f16(const float* __restrict__ in, _Float16* __restrict__ out, int n) {
  for (int t = blockIdx.x * blockDim.x + threadIdx.x; t < n;
       t += gridDim.x * blockDim.x)
    out[t] = (_Float16)in[t];
}

// xyz f32 [rows,3] -> f16 [rows, 32] zero-padded
__global__ void pad3_to_f16(const float* __restrict__ in, _Float16* __restrict__ out,
                            int rows) {
  const int total = rows * 32;
  for (int t = blockIdx.x * blockDim.x + threadIdx.x; t < total;
       t += gridDim.x * blockDim.x) {
    const int c = t & 31;
    const int r = t >> 5;
    out[t] = (c < 3) ? (_Float16)in[(size_t)r * 3 + c] : (_Float16)0.f;
  }
}

// STN fc3: trans[b,3,3] = h2[b,:]@W + bias + eye(3).  h2 f16 [16,256].
__global__ void stn_fc3(const _Float16* __restrict__ h,
                        const float* __restrict__ W,
                        const float* __restrict__ bias,
                        float* __restrict__ trans) {
  const int t = blockIdx.x * blockDim.x + threadIdx.x;
  if (t >= 16 * 9) return;
  const int b = t / 9, j = t % 9;
  float acc = bias[j];
  for (int k = 0; k < 256; ++k) acc += (float)h[b * 256 + k] * W[k * 9 + j];
  if ((j / 3) == (j % 3)) acc += 1.f;
  trans[t] = acc;
}

// x[b,n,d] = sum_c xyz[b,n,c] * trans[b,c,d]
__global__ void apply_trans(const float* __restrict__ xyz,
                            const float* __restrict__ trans,
                            float* __restrict__ xout, int Nn, int total) {
  for (int t = blockIdx.x * blockDim.x + threadIdx.x; t < total;
       t += gridDim.x * blockDim.x) {
    const int d = t % 3;
    const int bn = t / 3;
    const int b = bn / Nn;
    const float* T = trans + b * 9;
    const float* p = xyz + (size_t)bn * 3;
    xout[t] = p[0] * T[0 * 3 + d] + p[1] * T[1 * 3 + d] + p[2] * T[2 * 3 + d];
  }
}

// ---------------------------------------------------------------------------
// Farthest point sampling: one block per batch, sequential over S steps.
// Matches reference ordering: out[i] = far at loop entry (far_0 = 0),
// dist init 1e10, argmax ties -> lowest index.
// ---------------------------------------------------------------------------
__global__ void fps_kernel(const float* __restrict__ pts, int* __restrict__ fidx,
                           int Nn, int S) {
  const int b = blockIdx.x;
  const int tid = threadIdx.x;
  __shared__ float dist[4096];
  __shared__ float rv[256];
  __shared__ int ri[256];
  __shared__ int s_far;
  const float* P = pts + (size_t)b * Nn * 3;
  for (int n = tid; n < Nn; n += blockDim.x) dist[n] = 1e10f;
  if (tid == 0) s_far = 0;
  __syncthreads();
  for (int i = 0; i < S; ++i) {
    const int far = s_far;
    if (tid == 0) fidx[(size_t)b * S + i] = far;
    const float cx = P[far * 3 + 0], cy = P[far * 3 + 1], cz = P[far * 3 + 2];
    float bv = -1.f;
    int bi = 0;
    for (int n = tid; n < Nn; n += blockDim.x) {
      const float dx = P[n * 3 + 0] - cx;
      const float dy = P[n * 3 + 1] - cy;
      const float dz = P[n * 3 + 2] - cz;
      const float nd = fminf(dist[n], dx * dx + dy * dy + dz * dz);
      dist[n] = nd;
      if (nd > bv) { bv = nd; bi = n; }
    }
    rv[tid] = bv;
    ri[tid] = bi;
    __syncthreads();
    for (int s = blockDim.x >> 1; s > 0; s >>= 1) {
      if (tid < s) {
        const float ov = rv[tid + s];
        const int oi = ri[tid + s];
        if (ov > rv[tid] || (ov == rv[tid] && oi < ri[tid])) {
          rv[tid] = ov;
          ri[tid] = oi;
        }
      }
      __syncthreads();
    }
    if (tid == 0) s_far = ri[0];
    __syncthreads();
  }
}

// ---------------------------------------------------------------------------
// kNN: one block per (b,s) query; Kk sequential argmin passes over an LDS
// distance array (matches lax.top_k(-d, k): ascending d, ties lowest index).
// ---------------------------------------------------------------------------
__global__ void knn_kernel(const float* __restrict__ pts,
                           const float* __restrict__ qxyz,
                           int* __restrict__ nidx, int Nn, int S, int Kk) {
  const int q = blockIdx.x;  // global query id over B*S
  const int b = q / S;
  const int tid = threadIdx.x;
  __shared__ float dist[4096];
  __shared__ float rv[256];
  __shared__ int ri[256];
  const float qx = qxyz[q * 3 + 0], qy = qxyz[q * 3 + 1], qz = qxyz[q * 3 + 2];
  const float* P = pts + (size_t)b * Nn * 3;
  for (int n = tid; n < Nn; n += blockDim.x) {
    const float dx = P[n * 3 + 0] - qx;
    const float dy = P[n * 3 + 1] - qy;
    const float dz = P[n * 3 + 2] - qz;
    dist[n] = dx * dx + dy * dy + dz * dz;
  }
  __syncthreads();
  for (int j = 0; j < Kk; ++j) {
    float bv = 3.4e38f;
    int bi = Nn;
    for (int n = tid; n < Nn; n += blockDim.x) {
      const float d = dist[n];
      if (d < bv) { bv = d; bi = n; }
    }
    rv[tid] = bv;
    ri[tid] = bi;
    __syncthreads();
    for (int s = blockDim.x >> 1; s > 0; s >>= 1) {
      if (tid < s) {
        const float ov = rv[tid + s];
        const int oi = ri[tid + s];
        if (ov < rv[tid] || (ov == rv[tid] && oi < ri[tid])) {
          rv[tid] = ov;
          ri[tid] = oi;
        }
      }
      __syncthreads();
    }
    if (tid == 0) {
      nidx[(size_t)q * Kk + j] = ri[0];
      dist[ri[0]] = 3.4e38f;
    }
    __syncthreads();
  }
}

// ---------------------------------------------------------------------------
// Grouped MLP input (f16, padded rows): out[(b,s,k), 0:3] = pts[n]-newxyz[s],
// out[.., 3:3+Cf] = feats[n], out[.., 3+Cf:ldo] = 0.  n = nidx[b,s,k].
// ---------------------------------------------------------------------------
__global__ void group_gather(const float* __restrict__ pts,
                             const float* __restrict__ feats,
                             const float* __restrict__ newxyz,
                             const int* __restrict__ nidx,
                             _Float16* __restrict__ out,
                             int Nn, int S, int Kk, int Cf, int ldo, int total) {
  const int Cin = 3 + Cf;
  for (int t = blockIdx.x * blockDim.x + threadIdx.x; t < total;
       t += gridDim.x * blockDim.x) {
    const int c = t % ldo;
    const int r = t / ldo;
    float v = 0.f;
    if (c < Cin) {
      const int k = r % Kk;
      const int sq = r / Kk;  // global query id over B*S
      const int b = sq / S;
      const int n = nidx[(size_t)sq * Kk + k];
      if (c < 3)
        v = pts[((size_t)b * Nn + n) * 3 + c] - newxyz[(size_t)sq * 3 + c];
      else
        v = feats[((size_t)b * Nn + n) * Cf + (c - 3)];
    }
    out[(size_t)r * ldo + c] = (_Float16)v;
  }
}

// SA3 input: out[r, 0:3]=xyz, [3:3+Cf]=feats, pad to ldo with zeros (f16)
__global__ void concat_convert(const float* __restrict__ xyzf,
                               const float* __restrict__ feats,
                               _Float16* __restrict__ out, int rows, int Cf,
                               int ldo) {
  const int Cin = 3 + Cf;
  const int total = rows * ldo;
  for (int t = blockIdx.x * blockDim.x + threadIdx.x; t < total;
       t += gridDim.x * blockDim.x) {
    const int c = t % ldo;
    const int r = t / ldo;
    float v = 0.f;
    if (c < 3) v = xyzf[(size_t)r * 3 + c];
    else if (c < Cin) v = feats[(size_t)r * Cf + (c - 3)];
    out[t] = (_Float16)v;
  }
}

// newxyz[b,s,:] = pts[b, fidx[b,s], :]
__global__ void gather_xyz(const float* __restrict__ pts,
                           const int* __restrict__ fidx,
                           float* __restrict__ newxyz, int Nn, int S, int total) {
  for (int t = blockIdx.x * blockDim.x + threadIdx.x; t < total;
       t += gridDim.x * blockDim.x) {
    const int c = t % 3;
    const int s = t / 3;  // global query id over B*S
    const int b = s / S;
    newxyz[t] = pts[((size_t)b * Nn + fidx[s]) * 3 + c];
  }
}

// out[b] = dot(l3[b,:1024], Wf) + bf
__global__ void final_dot(const float* __restrict__ l3,
                          const float* __restrict__ Wf,
                          const float* __restrict__ bf, float* __restrict__ out) {
  const int b = blockIdx.x;
  const int tid = threadIdx.x;
  __shared__ float red[256];
  float acc = 0.f;
  for (int k = tid; k < 1024; k += blockDim.x)
    acc += l3[(size_t)b * 1024 + k] * Wf[k];
  red[tid] = acc;
  __syncthreads();
  for (int s = blockDim.x >> 1; s > 0; s >>= 1) {
    if (tid < s) red[tid] += red[tid + s];
    __syncthreads();
  }
  if (tid == 0) out[b] = red[0] + bf[0];
}

// ---------------------------------------------------------------------------
// Host orchestration
// ---------------------------------------------------------------------------
extern "C" void kernel_launch(void* const* d_in, const int* in_sizes, int n_in,
                              void* d_out, int out_size, void* d_ws, size_t ws_size,
                              hipStream_t stream) {
  (void)in_sizes; (void)n_in; (void)out_size; (void)ws_size;
  constexpr int Bn = 16, Np = 4096;
  constexpr int S1 = 512, K1 = 32;
  constexpr int S2 = 128, K2 = 64;

  const float* xyz = (const float*)d_in[0];
  // d_in[1]=faces, d_in[2]=idx : unused by the reference
  const float *Wsc1 = (const float*)d_in[3],  *bsc1 = (const float*)d_in[4];
  const float *Wsc2 = (const float*)d_in[5],  *bsc2 = (const float*)d_in[6];
  const float *Wsc3 = (const float*)d_in[7],  *bsc3 = (const float*)d_in[8];
  const float *Wf1 = (const float*)d_in[9],   *bf1 = (const float*)d_in[10];
  const float *Wf2 = (const float*)d_in[11],  *bf2 = (const float*)d_in[12];
  const float *Wf3 = (const float*)d_in[13],  *bf3 = (const float*)d_in[14];
  const float *Wa1 = (const float*)d_in[15],  *ba1 = (const float*)d_in[16];
  const float *Wa2 = (const float*)d_in[17],  *ba2 = (const float*)d_in[18];
  const float *Wa3 = (const float*)d_in[19],  *ba3 = (const float*)d_in[20];
  const float *Wb1 = (const float*)d_in[21],  *bb1 = (const float*)d_in[22];
  const float *Wb2 = (const float*)d_in[23],  *bb2 = (const float*)d_in[24];
  const float *Wb3 = (const float*)d_in[25],  *bb3 = (const float*)d_in[26];
  const float *Wc1 = (const float*)d_in[27],  *bc1 = (const float*)d_in[28];
  const float *Wc2 = (const float*)d_in[29],  *bc2 = (const float*)d_in[30];
  const float *Wc3 = (const float*)d_in[31],  *bc3 = (const float*)d_in[32];
  const float *Wfin = (const float*)d_in[33], *bfin = (const float*)d_in[34];
  float* out = (float*)d_out;

  // ---- workspace bump allocator (256B aligned) ----
  size_t off = 0;
  auto alloc = [&](size_t bytes) -> char* {
    char* p = (char*)d_ws + off;
    off += (bytes + 255) & ~(size_t)255;
    return p;
  };
  // f16 ping-pong activation buffers (sized for the largest user, padded lda)
  _Float16* buf0 = (_Float16*)alloc(sizeof(_Float16) * (size_t)20971520); // inputs (SA2: 131072x160)
  _Float16* buf1 = (_Float16*)alloc(sizeof(_Float16) * (size_t)16777216); // h1
  _Float16* buf2 = (_Float16*)alloc(sizeof(_Float16) * (size_t)16777216); // h2
  _Float16* wt   = (_Float16*)alloc(sizeof(_Float16) * (size_t)524288);   // transposed weight scratch
  float* stn_pool   = (float*)alloc(sizeof(float) * 16 * 1024);
  _Float16* pool16  = (_Float16*)alloc(sizeof(_Float16) * 16 * 1024);
  float* trans      = (float*)alloc(sizeof(float) * 16 * 9);
  float* xg         = (float*)alloc(sizeof(float) * (size_t)Bn * Np * 3);
  int*   fidx1      = (int*)alloc(sizeof(int) * Bn * S1);
  float* nxyz1      = (float*)alloc(sizeof(float) * Bn * S1 * 3);
  int*   nidx1      = (int*)alloc(sizeof(int) * (size_t)Bn * S1 * K1);
  float* l1f        = (float*)alloc(sizeof(float) * (size_t)Bn * S1 * 128);
  int*   fidx2      = (int*)alloc(sizeof(int) * Bn * S2);
  float* nxyz2      = (float*)alloc(sizeof(float) * Bn * S2 * 3);
  int*   nidx2      = (int*)alloc(sizeof(int) * (size_t)Bn * S2 * K2);
  float* l2f        = (float*)alloc(sizeof(float) * (size_t)Bn * S2 * 256);
  float* l3         = (float*)alloc(sizeof(float) * 16 * 1024);

  auto kpad = [](int K) { return (K + 31) & ~31; };

  auto gemm = [&](const _Float16* A, int lda, const float* W, const float* b,
                  _Float16* C, int ldc, int M, int N, int K) {
    const int Kp = kpad(K);
    weight_to_f16t<<<(N * Kp + 255) / 256, 256, 0, stream>>>(W, wt, K, N, Kp);
    const int strips = (M / 16) * (N / 64);
    gemm_wmma_f16<<<(strips + 7) / 8, 256, 0, stream>>>(A, lda, wt, b, C, ldc, M, N, Kp);
  };
  auto gemm_pool = [&](const _Float16* A, int lda, const float* W, const float* b,
                       float* pool, int M, int N, int K, int rps) {
    const int Kp = kpad(K);
    weight_to_f16t<<<(N * Kp + 255) / 256, 256, 0, stream>>>(W, wt, K, N, Kp);
    zero_f32<<<((M / rps) * N + 255) / 256, 256, 0, stream>>>(pool, (M / rps) * N);
    const int strips = (M / 16) * (N / 64);
    gemm_wmma_maxpool<<<(strips + 7) / 8, 256, 0, stream>>>(A, lda, wt, b, pool, M, N, Kp, rps);
  };

  // ================= STN =================
  const int MP = Bn * Np;  // 65536 points
  pad3_to_f16<<<(MP * 32 + 255) / 256, 256, 0, stream>>>(xyz, buf0, MP);
  gemm(buf0, 32, Wsc1, bsc1, buf1, 64, MP, 64, 3);
  gemm(buf1, 64, Wsc2, bsc2, buf2, 128, MP, 128, 64);
  gemm_pool(buf2, 128, Wsc3, bsc3, stn_pool, MP, 1024, 128, Np);   // -> [16,1024]
  f32_to_f16<<<(16 * 1024 + 255) / 256, 256, 0, stream>>>(stn_pool, pool16, 16 * 1024);
  gemm(pool16, 1024, Wf1, bf1, buf1, 512, 16, 512, 1024);
  gemm(buf1, 512, Wf2, bf2, buf2, 256, 16, 256, 512);
  stn_fc3<<<1, 256, 0, stream>>>(buf2, Wf3, bf3, trans);
  apply_trans<<<(MP * 3 + 255) / 256, 256, 0, stream>>>(xyz, trans, xg, Np, MP * 3);

  // ================= SA1 =================
  fps_kernel<<<Bn, 256, 0, stream>>>(xg, fidx1, Np, S1);
  gather_xyz<<<(Bn * S1 * 3 + 255) / 256, 256, 0, stream>>>(xg, fidx1, nxyz1, Np, S1, Bn * S1 * 3);
  knn_kernel<<<Bn * S1, 256, 0, stream>>>(xg, nxyz1, nidx1, Np, S1, K1);
  {
    const int rows = Bn * S1 * K1;  // 262144, Cin=6 -> lda 32
    const int tot = rows * 32;
    group_gather<<<(tot + 255) / 256, 256, 0, stream>>>(xg, xg, nxyz1, nidx1, buf0,
                                                        Np, S1, K1, 3, 32, tot);
    gemm(buf0, 32, Wa1, ba1, buf1, 64, rows, 64, 6);
    gemm(buf1, 64, Wa2, ba2, buf2, 64, rows, 64, 64);
    gemm_pool(buf2, 64, Wa3, ba3, l1f, rows, 128, 64, K1);          // -> [16*512,128]
  }

  // ================= SA2 =================
  fps_kernel<<<Bn, 256, 0, stream>>>(nxyz1, fidx2, S1, S2);
  gather_xyz<<<(Bn * S2 * 3 + 255) / 256, 256, 0, stream>>>(nxyz1, fidx2, nxyz2, S1, S2, Bn * S2 * 3);
  knn_kernel<<<Bn * S2, 256, 0, stream>>>(nxyz1, nxyz2, nidx2, S1, S2, K2);
  {
    const int rows = Bn * S2 * K2;  // 131072, Cin=131 -> lda 160
    const int tot = rows * 160;
    group_gather<<<(tot + 255) / 256, 256, 0, stream>>>(nxyz1, l1f, nxyz2, nidx2, buf0,
                                                        S1, S2, K2, 128, 160, tot);
    gemm(buf0, 160, Wb1, bb1, buf1, 128, rows, 128, 131);
    gemm(buf1, 128, Wb2, bb2, buf2, 128, rows, 128, 128);
    gemm_pool(buf2, 128, Wb3, bb3, l2f, rows, 256, 128, K2);        // -> [16*128,256]
  }

  // ================= SA3 (global) =================
  {
    const int rows = Bn * S2;  // 2048, Cin=259 -> lda 288
    concat_convert<<<(rows * 288 + 255) / 256, 256, 0, stream>>>(nxyz2, l2f, buf0, rows, 256, 288);
    gemm(buf0, 288, Wc1, bc1, buf1, 256, rows, 256, 259);
    gemm(buf1, 256, Wc2, bc2, buf2, 512, rows, 512, 256);
    gemm_pool(buf2, 512, Wc3, bc3, l3, rows, 1024, 512, S2);        // -> [16,1024]
  }

  // ================= head =================
  final_dot<<<Bn, 256, 0, stream>>>(l3, Wfin, bfin, out);
}
